// SentenceComparison_85667417686029
// MI455X (gfx1250) — compile-verified, matrix operand order checked by reading, not run on hardware
//
#include <hip/hip_runtime.h>

#define L_DIM 128
#define R_DIM 128
#define B_DIM 512
#define D_DIM 768
#define DK 32
#define LDA (DK + 2)

typedef __attribute__((ext_vector_type(2)))  float    v2f;
typedef __attribute__((ext_vector_type(8)))  float    v8f;
typedef __attribute__((ext_vector_type(16))) _Float16 v16h;

#if __has_builtin(__builtin_amdgcn_wmma_f32_16x16x4_f32)
#define USE_F32_WMMA 1
#else
#define USE_F32_WMMA 0
#endif

// Pass 1: inv_norm[l,d] = 1/sqrt(sum_b x[l,b,d]^2) for left then right,
// stored consecutively in workspace. Coalesced across d at each b step.
__global__ __launch_bounds__(256) void sc_norm_kernel(const float* __restrict__ left,
                                                      const float* __restrict__ right,
                                                      float* __restrict__ inv) {
  int g = blockIdx.x * 256 + threadIdx.x;          // 0 .. 2*L*D-1 (exact grid)
  const int total = L_DIM * D_DIM;
  const float* src = (g < total) ? left : right;
  int idx = (g < total) ? g : g - total;           // idx = l*D + d
  int l = idx / D_DIM;
  int d = idx - l * D_DIM;
  size_t base = (size_t)l * B_DIM * D_DIM + d;
  float s = 0.f;
  for (int b = 0; b < B_DIM; ++b) {
    float x = src[base + (size_t)b * D_DIM];
    s += x * x;
  }
  inv[g] = 1.0f / sqrtf(s);
}

// Pass 2: one block per batch b. Fused normalize+GEMM (WMMA) + mask +
// max/argmax + gathered differences.
__global__ __launch_bounds__(256) void sc_attn_kernel(const float* __restrict__ left,
                                                      const int*   __restrict__ llen_arr,
                                                      const float* __restrict__ right,
                                                      const int*   __restrict__ rlen_arr,
                                                      const float* __restrict__ inv,
                                                      float* __restrict__ out) {
  __shared__ alignas(16) unsigned char smem_raw[65536];
  float* smemF = (float*)smem_raw;
  float* As  = smemF;                // [128][LDA] stage (normalized left chunk)
  float* Bs  = smemF + 128 * LDA;    // [128][LDA] stage (normalized right chunk)
  float* att = smemF;                // [128][128] masked attention (aliases stage)
  int*   smemI = (int*)smem_raw;     // idx broadcast (aliases att, used after)

  const int b    = blockIdx.x;
  const int t    = threadIdx.x;
  const int lane = t & 31;
  const int wave = t >> 5;
  const int lm   = lane & 15;
  const int hi   = lane >> 4;
  const int m0   = wave * 16;

  const int Llen = llen_arr[b];
  const int Rlen = rlen_arr[b];

  const float* invL = inv;
  const float* invR = inv + L_DIM * D_DIM;

  const size_t off_ld = 0;
  const size_t off_rd = (size_t)L_DIM * B_DIM * D_DIM;
  const size_t off_lm = off_rd + (size_t)R_DIM * B_DIM * D_DIM;
  const size_t off_rm = off_lm + (size_t)B_DIM * L_DIM;
  const size_t off_ls = off_rm + (size_t)B_DIM * R_DIM;
  const size_t off_rs = off_ls + (size_t)L_DIM * B_DIM;

  v8f acc[8] = {};

  for (int dc = 0; dc < D_DIM; dc += DK) {
    __syncthreads();   // previous chunk's LDS reads complete before overwrite
    for (int i = t; i < 128 * DK; i += 256) {
      int r = i >> 5;             // DK == 32
      int k = i & (DK - 1);
      int d = dc + k;
      As[r * LDA + k] = left [((size_t)r * B_DIM + b) * D_DIM + d] * invL[r * D_DIM + d];
      Bs[r * LDA + k] = right[((size_t)r * B_DIM + b) * D_DIM + d] * invR[r * D_DIM + d];
    }
    __syncthreads();

#if USE_F32_WMMA
    // A 16x4 f32 frag: VGPR0=K(2*hi), VGPR1=K(2*hi+1), row = lane&15.
    // B 4x16 frag mirrors A with B[k][n] = Rv[n][k].
    for (int k = 0; k < DK; k += 4) {
      int ka = k + hi * 2;
      v2f a;
      a.x = As[(m0 + lm) * LDA + ka];
      a.y = As[(m0 + lm) * LDA + ka + 1];
#pragma unroll
      for (int n = 0; n < 8; ++n) {
        v2f bb;
        bb.x = Bs[(n * 16 + lm) * LDA + ka];
        bb.y = Bs[(n * 16 + lm) * LDA + ka + 1];
        acc[n] = __builtin_amdgcn_wmma_f32_16x16x4_f32(
            false, a, false, bb, (short)0, acc[n], false, false);
      }
    }
#else
    // Fallback: f16 inputs, f32 accumulate, K-step 32 == DK.
    {
      int kb = hi * 8;
      v16h a;
#pragma unroll
      for (int j = 0; j < 4; ++j) {
        a[2*j]     = (_Float16)As[(m0 + lm) * LDA + kb + 2*j];
        a[2*j + 1] = (_Float16)As[(m0 + lm) * LDA + kb + 2*j + 1];
        a[8 + 2*j]     = (_Float16)As[(m0 + lm) * LDA + 16 + kb + 2*j];
        a[8 + 2*j + 1] = (_Float16)As[(m0 + lm) * LDA + 16 + kb + 2*j + 1];
      }
#pragma unroll
      for (int n = 0; n < 8; ++n) {
        v16h bb;
#pragma unroll
        for (int j = 0; j < 4; ++j) {
          bb[2*j]     = (_Float16)Bs[(n*16 + lm) * LDA + kb + 2*j];
          bb[2*j + 1] = (_Float16)Bs[(n*16 + lm) * LDA + kb + 2*j + 1];
          bb[8 + 2*j]     = (_Float16)Bs[(n*16 + lm) * LDA + 16 + kb + 2*j];
          bb[8 + 2*j + 1] = (_Float16)Bs[(n*16 + lm) * LDA + 16 + kb + 2*j + 1];
        }
        acc[n] = __builtin_amdgcn_wmma_f32_16x16x32_f16(
            false, a, false, bb, (short)0, acc[n], false, false);
      }
    }
#endif
  }
  __syncthreads();   // all stage reads done; att may alias stage now

  // Apply masks and spill attention to LDS.
  // C layout: VGPR j -> M = m0 + j + 8*hi, N = 16n + (lane&15)
#pragma unroll
  for (int n = 0; n < 8; ++n) {
    int col = n * 16 + lm;
    float rmask = (col < Rlen) ? 1.f : 0.f;
#pragma unroll
    for (int j = 0; j < 8; ++j) {
      int row = m0 + hi * 8 + j;
      float lmask = (row < Llen) ? 1.f : 0.f;
      att[row * 128 + col] = acc[n][j] * lmask * rmask;
    }
  }
  __syncthreads();

  // Max + first-occurrence argmax; also emit masks and sims.
  int best_idx = 0;
  if (t < 128) {
    int l = t;
    float best = att[l * 128];
    for (int r = 1; r < 128; ++r) {
      float v = att[l * 128 + r];
      if (v > best) { best = v; best_idx = r; }
    }
    out[off_ls + (size_t)l * B_DIM + b] = best;             // left_sim.T [L,B]
    out[off_lm + (size_t)b * L_DIM + l] = (l < Llen) ? 1.f : 0.f;
  } else {
    int r = t - 128;
    float best = att[r];
    for (int l = 1; l < 128; ++l) {
      float v = att[l * 128 + r];
      if (v > best) { best = v; best_idx = l; }
    }
    out[off_rs + (size_t)r * B_DIM + b] = best;             // right_sim.T [R,B]
    out[off_rm + (size_t)b * R_DIM + r] = (r < Rlen) ? 1.f : 0.f;
  }
  __syncthreads();          // att reads complete
  smemI[t] = best_idx;      // t<128: l_idx[l]; t>=128: r_idx[r]
  __syncthreads();

  const int* lidx = smemI;
  const int* ridx = smemI + 128;

  // Gathered differences, coalesced over d.
  for (int i = t; i < L_DIM * D_DIM; i += 256) {
    int l = i / D_DIM;
    int d = i - l * D_DIM;
    float lv = left [((size_t)l       * B_DIM + b) * D_DIM + d];
    float rv = right[((size_t)lidx[l] * B_DIM + b) * D_DIM + d];
    float m  = (l < Llen) ? 1.f : 0.f;
    out[off_ld + ((size_t)l * B_DIM + b) * D_DIM + d] = (lv - rv) * m;
  }
  for (int i = t; i < R_DIM * D_DIM; i += 256) {
    int r = i / D_DIM;
    int d = i - r * D_DIM;
    float rv = right[((size_t)r       * B_DIM + b) * D_DIM + d];
    float lv = left [((size_t)ridx[r] * B_DIM + b) * D_DIM + d];
    float m  = (r < Rlen) ? 1.f : 0.f;
    out[off_rd + ((size_t)r * B_DIM + b) * D_DIM + d] = (rv - lv) * m;
  }
}

extern "C" void kernel_launch(void* const* d_in, const int* in_sizes, int n_in,
                              void* d_out, int out_size, void* d_ws, size_t ws_size,
                              hipStream_t stream) {
  const float* left  = (const float*)d_in[0];
  const int*   llen  = (const int*)  d_in[1];
  const float* right = (const float*)d_in[2];
  const int*   rlen  = (const int*)  d_in[3];
  float* out = (float*)d_out;
  float* inv = (float*)d_ws;   // needs 2*L*D floats = 786,432 bytes

  sc_norm_kernel<<<(2 * L_DIM * D_DIM) / 256, 256, 0, stream>>>(left, right, inv);
  sc_attn_kernel<<<B_DIM, 256, 0, stream>>>(left, llen, right, rlen, inv, out);
}
